// Memory_58059367908134
// MI455X (gfx1250) — matrix operand order repaired and checked by reading, not compiled, not run on hardware
//
#include <hip/hip_runtime.h>
#include <hip/hip_bf16.h>

#define B_  256
#define N_  32768
#define D_  2048
#define KK  8192
#define KSTEP 32

typedef __attribute__((ext_vector_type(16))) __bf16 v16bf;
typedef __attribute__((ext_vector_type(8)))  float  v8f;
typedef __attribute__((ext_vector_type(4)))  int    v4i;

typedef __attribute__((address_space(1))) v4i gas_v4i;   // global int4
typedef __attribute__((address_space(3))) v4i las_v4i;   // LDS int4

__device__ __forceinline__ gas_v4i* as_global(const void* p) {
    return (gas_v4i*)(unsigned long long)p;              // generic == global address
}
__device__ __forceinline__ las_v4i* as_lds(void* p) {
    return (las_v4i*)(unsigned)(unsigned long long)p;    // low 32 bits = LDS offset
}

#if __has_builtin(__builtin_amdgcn_global_load_async_to_lds_b128)
#define ASYNC_LDS 1
#else
#define ASYNC_LDS 0
#endif

__device__ __forceinline__ void wait_async0() {
#if __has_builtin(__builtin_amdgcn_s_wait_asynccnt)
    __builtin_amdgcn_s_wait_asynccnt(0);
#else
    asm volatile("s_wait_asynccnt 0x0" ::: "memory");
#endif
}

__device__ __forceinline__ unsigned short f32_to_bf16(float x) {
    unsigned u = __float_as_uint(x);
    unsigned r = (u + 0x7FFFu + ((u >> 16) & 1u)) >> 16;     // RNE (off critical path only)
    return (unsigned short)r;
}
// one v_perm_b32: [lo16 = a[31:16], hi16 = b[31:16]]
__device__ __forceinline__ unsigned pack_bf16(float a, float b) {
    return __builtin_amdgcn_perm(__float_as_uint(b), __float_as_uint(a), 0x07060302u);
}

__device__ __forceinline__ unsigned hash3(unsigned b, unsigned n, unsigned salt) {
    unsigned h = n * 2654435761u + b * 0x9E3779B9u + salt;
    h ^= h >> 16; h *= 0x85EBCA6Bu; h ^= h >> 13; h *= 0xC2B2AE35u; h ^= h >> 16;
    return h;
}

// ---------------- f32 -> bf16 conversion (f and f_nv, 1 MB each, L2 resident) --
__global__ void cvt_bf16(const float* __restrict__ src, unsigned short* __restrict__ dst, int n) {
    int i = blockIdx.x * blockDim.x + threadIdx.x;
    if (i < n) dst[i] = f32_to_bf16(src[i]);
}

// ---------------- fused dual GEMM: mat = f @ bank^T, mat_sim = f_nv @ bank^T ---
// grid 512, block 256 (8 waves). Block: full M=256, N-tile 64. Bank read ONCE from HBM.
// Double-buffered GLOBAL_LOAD_ASYNC_TO_LDS pipeline overlapped with WMMA.
__global__ void __launch_bounds__(256) gemm_dual(
    const float* __restrict__ bank, const unsigned short* __restrict__ fb,
    const unsigned short* __restrict__ gb,
    float* __restrict__ mat, float* __restrict__ msim)
{
    __shared__ __align__(16) unsigned short lf[2][256 * 32];   // f   tiles (bf16)
    __shared__ __align__(16) unsigned short lg[2][256 * 32];   // fnv tiles (bf16)
    __shared__ __align__(16) float          lb[2][64 * 32];    // bank tiles (f32)
    const int tid  = threadIdx.x;
    const int wave = tid >> 5, lane = tid & 31;
    const int n0   = blockIdx.x * 64;

    v8f accF[2][4], accG[2][4];
    #pragma unroll
    for (int a = 0; a < 2; a++)
        #pragma unroll
        for (int c = 0; c < 4; c++) { accF[a][c] = (v8f)0.0f; accG[a][c] = (v8f)0.0f; }

#if ASYNC_LDS
    auto issue_tiles = [&](int k0, int s) {
        #pragma unroll
        for (int p = 0; p < 4; p++) {                      // 1024 b128 per operand tile
            int i = tid + p * 256;
            int r = i >> 2, c = i & 3;
            __builtin_amdgcn_global_load_async_to_lds_b128(
                as_global(fb + r * D_ + k0 + c * 8),
                as_lds((char*)lf[s] + r * 64 + c * 16), 0, 0);
            __builtin_amdgcn_global_load_async_to_lds_b128(
                as_global(gb + r * D_ + k0 + c * 8),
                as_lds((char*)lg[s] + r * 64 + c * 16), 0, 0);
        }
        #pragma unroll
        for (int p = 0; p < 2; p++) {                      // 512 b128 bank tile (f32)
            int i = tid + p * 256;
            int r = i >> 3, c = i & 7;
            __builtin_amdgcn_global_load_async_to_lds_b128(
                as_global(bank + (long)(n0 + r) * D_ + k0 + c * 4),
                as_lds((char*)lb[s] + r * 128 + c * 16), 0, 0);
        }
    };
    issue_tiles(0, 0);
    wait_async0();
    __syncthreads();
    int buf = 0;
#else
    const int buf = 0;
#endif

    for (int k0 = 0; k0 < D_; k0 += KSTEP) {
#if ASYNC_LDS
        if (k0 + KSTEP < D_) issue_tiles(k0 + KSTEP, buf ^ 1);   // prefetch next tile (DMA)
#else
        // batched fallback: all loads into regs, then all LDS stores
        uint4 tf[4], tg[4], tb[2];
        #pragma unroll
        for (int p = 0; p < 4; p++) {
            int i = tid + p * 256;
            int r = i >> 2, c = i & 3;
            tf[p] = *((const uint4*)(fb + r * D_ + k0) + c);
            tg[p] = *((const uint4*)(gb + r * D_ + k0) + c);
        }
        #pragma unroll
        for (int p = 0; p < 2; p++) {
            int i = tid + p * 256;
            int r = i >> 3, c = i & 7;
            tb[p] = *((const uint4*)(bank + (long)(n0 + r) * D_ + k0) + c);
        }
        if (k0 > 0) __syncthreads();
        #pragma unroll
        for (int p = 0; p < 4; p++) {
            int i = tid + p * 256;
            ((uint4*)lf[0])[i] = tf[p];
            ((uint4*)lg[0])[i] = tg[p];
        }
        #pragma unroll
        for (int p = 0; p < 2; p++) {
            int i = tid + p * 256;
            ((uint4*)lb[0])[i] = tb[p];
        }
        __syncthreads();
#endif

        // ---- fragments (contiguous K order, lane half h = K 16h..16h+15 for A and B; B per-lane = N column)
        union Frag { v16bf v; uint4 q[2]; unsigned u[8]; };
        Frag fa[2], ga[2], bfr[4];
        const int ar   = lane & 15;
        const int aoff = (lane >> 4) * 32;                 // bytes: K0-15 vs K16-31
        #pragma unroll
        for (int mi = 0; mi < 2; mi++) {
            const char* pf = (const char*)lf[buf] + (wave * 32 + mi * 16 + ar) * 64 + aoff;
            fa[mi].q[0] = *(const uint4*)pf;   fa[mi].q[1] = *(const uint4*)(pf + 16);
            const char* pg = (const char*)lg[buf] + (wave * 32 + mi * 16 + ar) * 64 + aoff;
            ga[mi].q[0] = *(const uint4*)pg;   ga[mi].q[1] = *(const uint4*)(pg + 16);
        }
        const int koff = (lane >> 4) * 64;                 // f32 bytes: K0-15 vs K16-31
        #pragma unroll
        for (int nc = 0; nc < 4; nc++) {
            const float4* pb = (const float4*)((const char*)lb[buf] + (nc * 16 + ar) * 128 + koff);
            float4 x0 = pb[0], x1 = pb[1], x2 = pb[2], x3 = pb[3];
            bfr[nc].u[0] = pack_bf16(x0.x, x0.y);  bfr[nc].u[1] = pack_bf16(x0.z, x0.w);
            bfr[nc].u[2] = pack_bf16(x1.x, x1.y);  bfr[nc].u[3] = pack_bf16(x1.z, x1.w);
            bfr[nc].u[4] = pack_bf16(x2.x, x2.y);  bfr[nc].u[5] = pack_bf16(x2.z, x2.w);
            bfr[nc].u[6] = pack_bf16(x3.x, x3.y);  bfr[nc].u[7] = pack_bf16(x3.z, x3.w);
        }
        #pragma unroll
        for (int mi = 0; mi < 2; mi++)
            #pragma unroll
            for (int nc = 0; nc < 4; nc++) {
                accF[mi][nc] = __builtin_amdgcn_wmma_f32_16x16x32_bf16(
                    false, fa[mi].v, false, bfr[nc].v, (short)0, accF[mi][nc], false, false);
                accG[mi][nc] = __builtin_amdgcn_wmma_f32_16x16x32_bf16(
                    false, ga[mi].v, false, bfr[nc].v, (short)0, accG[mi][nc], false, false);
            }
#if ASYNC_LDS
        wait_async0();          // next tile's DMA done (overlapped with WMMA above)
        __syncthreads();        // all waves done reading current buffer
        buf ^= 1;
#endif
    }

    // ---- epilogue: C layout — VGPR r: lanes0-15 M=r, lanes16-31 M=8+r; lane%16 = N
    const int colBase = n0 + (lane & 15);
    const int rowHalf = (lane >> 4) * 8;
    #pragma unroll
    for (int mi = 0; mi < 2; mi++)
        #pragma unroll
        for (int nc = 0; nc < 4; nc++)
            #pragma unroll
            for (int r = 0; r < 8; r++) {
                int row = wave * 32 + mi * 16 + rowHalf + r;
                int col = colBase + nc * 16;
                mat [row * N_ + col] = accF[mi][nc][r];
                msim[row * N_ + col] = accG[mi][nc][r];
            }
}

// ---------------- l_pos[b] = <f_nv[b], bank[indexes[b]]> ------------------------
__global__ void lpos_kernel(const float* __restrict__ fnv, const float* __restrict__ bank,
                            const int* __restrict__ indexes, float* __restrict__ lpos) {
    int b = blockIdx.x, t = threadIdx.x;
    __shared__ float sf[256];
    const float* a = fnv + b * D_;
    const float* k = bank + (long)indexes[b] * D_;
    float s = 0.f;
    for (int j = t; j < D_; j += 256) s += a[j] * k[j];
    sf[t] = s; __syncthreads();
    for (int st = 128; st > 0; st >>= 1) { if (t < st) sf[t] += sf[t + st]; __syncthreads(); }
    if (t == 0) lpos[b] = sf[0];
}

// ---------------- per-row sampling + 3x CE(target 0) ---------------------------
__global__ void __launch_bounds__(256) loss_rows(
    const float* __restrict__ mat, const float* __restrict__ msim,
    const int* __restrict__ labels, const int* __restrict__ indexes,
    const float* __restrict__ lpos, float* __restrict__ rloss)
{
    const int b = blockIdx.x, t = threadIdx.x;
    __shared__ int   hist[256];
    __shared__ int   pscan[256];
    __shared__ float sf[256];
    __shared__ unsigned sk[256];
    __shared__ int   si[256];
    __shared__ int   blab, cL1, cL2, need1, need2, idxA, idxB;
    __shared__ float sMneg;

    if (t == 0) blab = labels[indexes[b]];
    hist[t] = 0;
    __syncthreads();

    const int n0 = t * 128;
    unsigned bk1 = 0; int bi1 = -1; unsigned bk2 = 0; int bi2 = -1;
    for (int i = 0; i < 128; i++) {
        int n = n0 + i;
        if (labels[n] == blab) {
            unsigned k1 = hash3(b, n, 0x1111u); if (bi1 < 0 || k1 > bk1) { bk1 = k1; bi1 = n; }
            unsigned k2 = hash3(b, n, 0x2222u); if (bi2 < 0 || k2 > bk2) { bk2 = k2; bi2 = n; }
        } else {
            unsigned k3 = hash3(b, n, 0x3333u);
            atomicAdd(&hist[k3 >> 24], 1);
        }
    }
    sk[t] = bk1; si[t] = bi1; __syncthreads();
    for (int s = 128; s > 0; s >>= 1) {
        if (t < s) { bool take = (si[t] < 0) || (si[t+s] >= 0 && sk[t+s] > sk[t]);
                     if (take) { sk[t] = sk[t+s]; si[t] = si[t+s]; } }
        __syncthreads();
    }
    if (t == 0) idxA = si[0];
    __syncthreads();
    sk[t] = bk2; si[t] = bi2; __syncthreads();
    for (int s = 128; s > 0; s >>= 1) {
        if (t < s) { bool take = (si[t] < 0) || (si[t+s] >= 0 && sk[t+s] > sk[t]);
                     if (take) { sk[t] = sk[t+s]; si[t] = si[t+s]; } }
        __syncthreads();
    }
    if (t == 0) {
        idxB = si[0];
        int cum = 0, c = 255, nd = KK;
        for (int j = 0; j < 256; j++) { if (cum + hist[j] > KK) { c = j; nd = KK - cum; break; } cum += hist[j]; }
        cL1 = c; need1 = nd;
    }
    __syncthreads();
    hist[t] = 0; __syncthreads();
    for (int i = 0; i < 128; i++) {
        int n = n0 + i; if (labels[n] == blab) continue;
        unsigned k3 = hash3(b, n, 0x3333u);
        if ((int)(k3 >> 24) == cL1) atomicAdd(&hist[(k3 >> 16) & 255], 1);
    }
    __syncthreads();
    if (t == 0) {
        int cum = 0, c = 255, nd = need1;
        for (int j = 0; j < 256; j++) { if (cum + hist[j] > need1) { c = j; nd = need1 - cum; break; } cum += hist[j]; }
        cL2 = c; need2 = nd;
    }
    __syncthreads();
    const unsigned T16 = ((unsigned)cL1 << 8) | (unsigned)cL2;
    int bc = 0;
    for (int i = 0; i < 128; i++) {
        int n = n0 + i; if (labels[n] == blab) continue;
        if ((hash3(b, n, 0x3333u) >> 16) == T16) bc++;
    }
    pscan[t] = bc; __syncthreads();
    if (t == 0) { int acc = 0; for (int j = 0; j < 256; j++) { int v = pscan[j]; pscan[j] = acc; acc += v; } }
    __syncthreads();

    const float invT = 1.0f / 0.07f;
    const float* row = msim + b * N_;
    float mx = -3.0e38f;
    {
        int rk = pscan[t];
        for (int i = 0; i < 128; i++) {
            int n = n0 + i; if (labels[n] == blab) continue;
            unsigned k16 = hash3(b, n, 0x3333u) >> 16;
            bool sel = (k16 < T16) || (k16 == T16 && (rk++) < need2);
            if (sel) mx = fmaxf(mx, row[n] * invT);
        }
    }
    sf[t] = mx; __syncthreads();
    for (int s = 128; s > 0; s >>= 1) { if (t < s) sf[t] = fmaxf(sf[t], sf[t+s]); __syncthreads(); }
    if (t == 0) sMneg = sf[0];
    __syncthreads();
    const float Mneg = sMneg;
    float sm = 0.f;
    {
        int rk = pscan[t];
        for (int i = 0; i < 128; i++) {
            int n = n0 + i; if (labels[n] == blab) continue;
            unsigned k16 = hash3(b, n, 0x3333u) >> 16;
            bool sel = (k16 < T16) || (k16 == T16 && (rk++) < need2);
            if (sel) sm += __expf(row[n] * invT - Mneg);
        }
    }
    sf[t] = sm; __syncthreads();
    for (int s = 128; s > 0; s >>= 1) { if (t < s) sf[t] += sf[t+s]; __syncthreads(); }
    if (t == 0) {
        float Sneg = sf[0];
        float ps[3] = { lpos[b], msim[b * N_ + idxB], mat[b * N_ + idxA] };
        float total = 0.f;
        for (int j = 0; j < 3; j++) {
            float pl = ps[j] * invT;
            float M  = fmaxf(Mneg, pl);
            float lse = M + __logf(__expf(Mneg - M) * Sneg + __expf(pl - M));
            total += lse - pl;
        }
        rloss[b] = total * (1.0f / (float)B_);
    }
}

__global__ void reduce_loss(const float* __restrict__ rloss, float* __restrict__ out) {
    int t = threadIdx.x;
    __shared__ float sf[256];
    sf[t] = rloss[t]; __syncthreads();
    for (int s = 128; s > 0; s >>= 1) { if (t < s) sf[t] += sf[t+s]; __syncthreads(); }
    if (t == 0) out[0] = sf[0];
}

// ---------------- new_features = features; rows[indexes] = normalize(momentum upd)
__global__ void copy_bank(const float* __restrict__ src, float* __restrict__ dst) {
    long i = (long)blockIdx.x * blockDim.x + threadIdx.x;   // one float4 per thread
    float4 v = ((const float4*)src)[i];
    long o = i * 4;
    dst[o] = v.x; dst[o+1] = v.y; dst[o+2] = v.z; dst[o+3] = v.w;
}

__global__ void update_rows(const float* __restrict__ bank, const float* __restrict__ f,
                            const int* __restrict__ indexes, float* __restrict__ dst) {
    int b = blockIdx.x, t = threadIdx.x;
    __shared__ float sf[256];
    __shared__ float srn;
    int idx = indexes[b];
    const float* kr = bank + (long)idx * D_;
    const float* fr = f + b * D_;
    float ss = 0.f;
    for (int j = t; j < D_; j += 256) { float u = 0.2f * kr[j] + 0.8f * fr[j]; ss += u * u; }
    sf[t] = ss; __syncthreads();
    for (int s = 128; s > 0; s >>= 1) { if (t < s) sf[t] += sf[t+s]; __syncthreads(); }
    if (t == 0) srn = 1.0f / fmaxf(sqrtf(sf[0]), 1e-12f);
    __syncthreads();
    float rn = srn;
    float* o = dst + (long)idx * D_;
    for (int j = t; j < D_; j += 256) { float u = 0.2f * kr[j] + 0.8f * fr[j]; o[j] = u * rn; }
}

extern "C" void kernel_launch(void* const* d_in, const int* in_sizes, int n_in,
                              void* d_out, int out_size, void* d_ws, size_t ws_size,
                              hipStream_t stream) {
    (void)in_sizes; (void)n_in; (void)out_size; (void)ws_size;
    const float* f        = (const float*)d_in[0];
    const float* f_nv     = (const float*)d_in[1];
    const float* features = (const float*)d_in[2];
    const int*   labels   = (const int*)d_in[3];
    const int*   indexes  = (const int*)d_in[4];
    float* out = (float*)d_out;

    char* ws = (char*)d_ws;
    unsigned short* fb  = (unsigned short*)ws;                       // 1 MB
    unsigned short* gb  = (unsigned short*)(ws + (1u << 20));        // 1 MB
    float* mat   = (float*)(ws + (2u << 20));                        // 32 MB
    float* msim  = (float*)(ws + (2u << 20) + ((size_t)32 << 20));   // 32 MB
    float* lpos  = (float*)(ws + (2u << 20) + ((size_t)64 << 20));
    float* rloss = lpos + B_;

    cvt_bf16<<<(B_ * D_ + 255) / 256, 256, 0, stream>>>(f,    fb, B_ * D_);
    cvt_bf16<<<(B_ * D_ + 255) / 256, 256, 0, stream>>>(f_nv, gb, B_ * D_);
    gemm_dual<<<N_ / 64, 256, 0, stream>>>(features, fb, gb, mat, msim);
    lpos_kernel<<<B_, 256, 0, stream>>>(f_nv, features, indexes, lpos);
    loss_rows<<<B_, 256, 0, stream>>>(mat, msim, labels, indexes, lpos, rloss);
    reduce_loss<<<1, 256, 0, stream>>>(rloss, out);
    copy_bank<<<(N_ * (long)D_) / (256 * 4), 256, 0, stream>>>(features, out + 1);
    update_rows<<<B_, 256, 0, stream>>>(features, f, indexes, out + 1);
}
